// NetVLAD_71700184039861
// MI455X (gfx1250) — matrix operand order repaired and checked by reading, not compiled, not run on hardware
//
#include <hip/hip_runtime.h>
#include <hip/hip_bf16.h>

typedef __attribute__((ext_vector_type(16))) _Float16 v16h;
typedef __attribute__((ext_vector_type(8)))  _Float16 v8h;
typedef __attribute__((ext_vector_type(8)))  float    v8f;

#define C_DIM 512
#define S_DIM 784
#define SP    800      // S padded to multiple of 32 for GEMM2 K-dim
#define K_DIM 64
#define N_IMG 64

static __device__ __forceinline__ v16h make_frag(const _Float16* lo, const _Float16* hi) {
    v16h f;
    ((v8h*)&f)[0] = *(const v8h*)lo;
    ((v8h*)&f)[1] = *(const v8h*)hi;
    return f;
}

static __device__ __forceinline__ v8f wmma_f16(v16h a, v16h b, v8f c) {
    return __builtin_amdgcn_wmma_f32_16x16x32_f16(false, a, false, b, (short)0, c, false, false);
}

// ---------------------------------------------------------------------------
// Kernel 0: convert conv_w f32 -> f16, zero a_sum + norm2 accumulators
// ---------------------------------------------------------------------------
__global__ void prep_kernel(const float* __restrict__ conv_w,
                            _Float16* __restrict__ convw_h,
                            float* __restrict__ zbuf) {
    int i = blockIdx.x * 256 + threadIdx.x;
    if (i < K_DIM * C_DIM) convw_h[i] = (_Float16)conv_w[i];
    if (i < K_DIM * N_IMG + N_IMG) zbuf[i] = 0.0f;   // a_sum[64*64] + norm2[64]
}

// ---------------------------------------------------------------------------
// Kernel 1: x f32[n][c][s] -> x_h f16[n][c][s_pad] (zero padded) and
//           xT_h f16[n][s][c] via LDS 32x32 tile transpose. Coalesced both ways.
// grid (25, 16, 64), block (32, 8)
// ---------------------------------------------------------------------------
__global__ void convert_x_kernel(const float* __restrict__ x,
                                 _Float16* __restrict__ x_h,
                                 _Float16* __restrict__ xT_h) {
    __shared__ float tile[32][33];
    int n  = blockIdx.z;
    int c0 = blockIdx.y * 32;
    int s0 = blockIdx.x * 32;
    int tx = threadIdx.x;   // 0..31
    int ty = threadIdx.y;   // 0..7
    const float* xn = x + (size_t)n * C_DIM * S_DIM;
#pragma unroll
    for (int j = 0; j < 4; ++j) {
        int cl = ty + j * 8;
        int s  = s0 + tx;                      // s < 800 always
        float v = (s < S_DIM) ? xn[(size_t)(c0 + cl) * S_DIM + s] : 0.0f;
        tile[cl][tx] = v;
        x_h[((size_t)n * C_DIM + (c0 + cl)) * SP + s] = (_Float16)v;
    }
    __syncthreads();
#pragma unroll
    for (int j = 0; j < 4; ++j) {
        int sl = ty + j * 8;
        int s  = s0 + sl;
        if (s < S_DIM)
            xT_h[((size_t)n * S_DIM + s) * C_DIM + (c0 + tx)] = (_Float16)tile[tx][sl];
    }
}

// ---------------------------------------------------------------------------
// Kernel 2: GEMM1 (logits^T = x^T * conv_w^T) + bias + softmax over k,
//           a_sum accumulation, store a as f16 [n][k][s_pad] via LDS transpose.
// grid (7, 64), block 256 (8 waves). Wave w handles s-tile st = bx*8+w (49 tiles).
// ---------------------------------------------------------------------------
__global__ void gemm1_softmax_kernel(const float* __restrict__ conv_b,
                                     const _Float16* __restrict__ convw_h,
                                     const _Float16* __restrict__ xT_h,
                                     _Float16* __restrict__ a_h,
                                     float* __restrict__ a_sum) {
    __shared__ _Float16 lds[8][1024];         // per-wave 16(s) x 64(k) tile, k-major
    int wave = threadIdx.x >> 5;
    int lane = threadIdx.x & 31;
    int l16  = lane & 15;
    int kg   = lane >> 4;
    int n    = blockIdx.y;
    int st   = blockIdx.x * 8 + wave;
    _Float16* a_n = a_h + (size_t)n * K_DIM * SP;

    if (st == 49) {                            // zero the s-pad region [784,800)
        v8h z;
#pragma unroll
        for (int i = 0; i < 8; ++i) z[i] = (_Float16)0;
#pragma unroll
        for (int r = 0; r < 2; ++r) {
            _Float16* p = a_n + (size_t)(lane + r * 32) * SP + S_DIM;
            ((v8h*)p)[0] = z;
            ((v8h*)p)[1] = z;
        }
        return;
    }
    if (st >= 49) return;
    int s0 = st * 16;

    v8f acc[4];
#pragma unroll
    for (int t = 0; t < 4; ++t)
#pragma unroll
        for (int i = 0; i < 8; ++i) acc[t][i] = 0.0f;

    // A = xT tile: row m=s-local (l16), K-dim = c. Contiguous k = 16*kg + s, same order as B.
    const _Float16* aBase = xT_h + ((size_t)n * S_DIM + (s0 + l16)) * C_DIM + 16 * kg;
    for (int cs = 0; cs < 16; ++cs) {
        int c0 = cs * 32;
        v16h afrag = make_frag(aBase + c0, aBase + c0 + 8);
#pragma unroll
        for (int t = 0; t < 4; ++t) {
            // B = conv_w^T tile: col n=k (t*16+l16), rows = c contiguous (16 halves)
            const _Float16* bp = convw_h + (size_t)(t * 16 + l16) * C_DIM + c0 + kg * 16;
            v16h bfrag = make_frag(bp, bp + 8);
            acc[t] = wmma_f16(afrag, bfrag, acc[t]);
        }
    }

    // bias: column k = t*16 + l16
#pragma unroll
    for (int t = 0; t < 4; ++t) {
        float b = conv_b[t * 16 + l16];
#pragma unroll
        for (int i = 0; i < 8; ++i) acc[t][i] += b;
    }

    // softmax over k (row m = v + 8*kg lives across the 16-lane group x 4 tiles)
#pragma unroll
    for (int v = 0; v < 8; ++v) {
        float mx = fmaxf(fmaxf(acc[0][v], acc[1][v]), fmaxf(acc[2][v], acc[3][v]));
#pragma unroll
        for (int mask = 1; mask < 16; mask <<= 1)
            mx = fmaxf(mx, __shfl_xor(mx, mask, 32));
        float sum = 0.0f;
#pragma unroll
        for (int t = 0; t < 4; ++t) {
            acc[t][v] = __expf(acc[t][v] - mx);
            sum += acc[t][v];
        }
#pragma unroll
        for (int mask = 1; mask < 16; mask <<= 1)
            sum += __shfl_xor(sum, mask, 32);
        float r = 1.0f / sum;
#pragma unroll
        for (int t = 0; t < 4; ++t) acc[t][v] *= r;
    }

    // a_sum[k] += column sums (this s-tile's contribution)
#pragma unroll
    for (int t = 0; t < 4; ++t) {
        float cs_ = 0.0f;
#pragma unroll
        for (int v = 0; v < 8; ++v) cs_ += acc[t][v];
        cs_ += __shfl_xor(cs_, 16, 32);
        if (kg == 0) atomicAdd(&a_sum[n * K_DIM + t * 16 + l16], cs_);
    }

    // transpose (s,k) -> (k,s) through per-wave LDS, then coalesced f16 stores
#pragma unroll
    for (int t = 0; t < 4; ++t)
#pragma unroll
        for (int v = 0; v < 8; ++v) {
            int k = t * 16 + l16;
            int m = v + 8 * kg;
            lds[wave][k * 16 + m] = (_Float16)acc[t][v];
        }
    asm volatile("s_wait_dscnt 0" ::: "memory");   // per-wave LDS write->read ordering
#pragma unroll
    for (int r = 0; r < 2; ++r) {
        int k = lane + r * 32;
        const v8h* p = (const v8h*)&lds[wave][k * 16];
        _Float16* g = a_n + (size_t)k * SP + s0;
        ((v8h*)g)[0] = p[0];
        ((v8h*)g)[1] = p[1];
    }
}

// ---------------------------------------------------------------------------
// Kernel 3: GEMM2 ax = a(64 x 800) * x^T(800 x C-slice), subtract
//           a_sum*centroids, write unnormalized vlad, accumulate norm^2.
// grid (4, 64), block 256 (8 waves). Wave owns 16 c-columns, all 64 k.
// ---------------------------------------------------------------------------
__global__ void gemm2_vlad_kernel(const _Float16* __restrict__ a_h,
                                  const _Float16* __restrict__ x_h,
                                  const float* __restrict__ a_sum,
                                  const float* __restrict__ cent,
                                  float* __restrict__ out,
                                  float* __restrict__ norm2) {
    int wave = threadIdx.x >> 5;
    int lane = threadIdx.x & 31;
    int l16  = lane & 15;
    int kg   = lane >> 4;
    int n    = blockIdx.y;
    int c0   = blockIdx.x * 128 + wave * 16;

    v8f acc[4];
#pragma unroll
    for (int t = 0; t < 4; ++t)
#pragma unroll
        for (int i = 0; i < 8; ++i) acc[t][i] = 0.0f;

    const _Float16* aN    = a_h + (size_t)n * K_DIM * SP;
    const _Float16* bBase = x_h + ((size_t)n * C_DIM + (c0 + l16)) * SP + kg * 16;

    for (int ssi = 0; ssi < 25; ++ssi) {
        int s0 = ssi * 32;
        v16h bfrag = make_frag(bBase + s0, bBase + s0 + 8);  // 16 contiguous halves
#pragma unroll
        for (int t = 0; t < 4; ++t) {
            const _Float16* ap = aN + (size_t)(t * 16 + l16) * SP + s0 + 16 * kg;
            v16h afrag = make_frag(ap, ap + 8);
            acc[t] = wmma_f16(afrag, bfrag, acc[t]);
        }
    }

    int c = c0 + l16;
    float ssq = 0.0f;
#pragma unroll
    for (int t = 0; t < 4; ++t) {
#pragma unroll
        for (int v = 0; v < 8; ++v) {
            int k = t * 16 + v + 8 * kg;      // D row m = v + 8*kg
            float val = acc[t][v] - a_sum[n * K_DIM + k] * cent[(size_t)k * C_DIM + c];
            out[(size_t)n * (K_DIM * C_DIM) + (size_t)k * C_DIM + c] = val;
            ssq += val * val;
        }
    }
#pragma unroll
    for (int mask = 1; mask < 32; mask <<= 1) ssq += __shfl_xor(ssq, mask, 32);
    if (lane == 0) atomicAdd(&norm2[n], ssq);
}

// ---------------------------------------------------------------------------
// Kernel 4: global L2 normalize per image
// ---------------------------------------------------------------------------
__global__ void normalize_kernel(float* __restrict__ out,
                                 const float* __restrict__ norm2) {
    size_t i = (size_t)blockIdx.x * 256 + threadIdx.x;
    int n = (int)(i >> 15);                   // 32768 elements per image
    float nrm = fmaxf(sqrtf(norm2[n]), 1e-12f);
    out[i] *= (1.0f / nrm);
}

// ---------------------------------------------------------------------------
extern "C" void kernel_launch(void* const* d_in, const int* in_sizes, int n_in,
                              void* d_out, int out_size, void* d_ws, size_t ws_size,
                              hipStream_t stream) {
    const float* x      = (const float*)d_in[0];   // [64,512,28,28]
    const float* conv_w = (const float*)d_in[1];   // [64,512]
    const float* conv_b = (const float*)d_in[2];   // [64]
    const float* cent   = (const float*)d_in[3];   // [64,512]
    float* out = (float*)d_out;                    // [64, 32768]

    // workspace layout (bytes)
    char* ws = (char*)d_ws;
    const size_t OFF_CONVW = 0;                                        // 64*512*2      = 64 KB
    const size_t OFF_XH    = OFF_CONVW + (size_t)K_DIM * C_DIM * 2;    // 64*512*800*2  = 50 MB
    const size_t OFF_XT    = OFF_XH + (size_t)N_IMG * C_DIM * SP * 2;  // 64*784*512*2  = 49 MB
    const size_t OFF_AH    = OFF_XT + (size_t)N_IMG * S_DIM * C_DIM * 2; // 64*64*800*2 = 6.25 MB
    const size_t OFF_ASUM  = OFF_AH + (size_t)N_IMG * K_DIM * SP * 2;  // 64*64*4
    const size_t OFF_NORM2 = OFF_ASUM + (size_t)N_IMG * K_DIM * 4;     // 64*4 (contiguous after a_sum)

    _Float16* convw_h = (_Float16*)(ws + OFF_CONVW);
    _Float16* x_h     = (_Float16*)(ws + OFF_XH);
    _Float16* xT_h    = (_Float16*)(ws + OFF_XT);
    _Float16* a_h     = (_Float16*)(ws + OFF_AH);
    float*    a_sum   = (float*)(ws + OFF_ASUM);
    float*    norm2   = (float*)(ws + OFF_NORM2);

    prep_kernel<<<128, 256, 0, stream>>>(conv_w, convw_h, a_sum /* zbuf covers a_sum+norm2 */);
    convert_x_kernel<<<dim3(25, 16, 64), dim3(32, 8), 0, stream>>>(x, x_h, xT_h);
    gemm1_softmax_kernel<<<dim3(7, 64), 256, 0, stream>>>(conv_b, convw_h, xT_h, a_h, a_sum);
    gemm2_vlad_kernel<<<dim3(4, 64), 256, 0, stream>>>(a_h, x_h, a_sum, cent, out, norm2);
    normalize_kernel<<<8192, 256, 0, stream>>>(out, norm2);
}
